// SetTransformer_43181601194590
// MI455X (gfx1250) — matrix operand order repaired and checked by reading, not compiled, not run on hardware
//
#include <hip/hip_runtime.h>
#include <cmath>

// ---------------- model dims ----------------
#define BB   8
#define NN   1024
#define DIN_ 128
#define DH_  2048
#define DL_  1024
#define NC_  18
#define NI_  256
#define HH   8

// ---------------- WMMA types ----------------
typedef __attribute__((ext_vector_type(16))) __bf16 v16bf;
typedef __attribute__((ext_vector_type(8)))  float  v8f;

union Frag { unsigned u[8]; v16bf v; };

// packed f32x2 -> bf16x2 conversion (RNE)
__device__ __forceinline__ unsigned packbf(float a, float b) {
#if __has_builtin(__builtin_amdgcn_cvt_pk_bf16_f32)
    auto r = __builtin_amdgcn_cvt_pk_bf16_f32(a, b);
    union { decltype(r) v; unsigned u; } p;
    p.v = r;
    return p.u;
#else
    union { __bf16 h[2]; unsigned u; } p;
    p.h[0] = (__bf16)a;
    p.h[1] = (__bf16)b;
    return p.u;
#endif
}
__device__ __forceinline__ v8f wmma_bf(const Frag& a, const Frag& b, v8f c) {
    return __builtin_amdgcn_wmma_f32_16x16x32_bf16(false, a.v, false, b.v,
                                                   (short)0, c, false, false);
}
// async 16B global -> LDS copy (per active lane), tracked by ASYNCcnt
__device__ __forceinline__ void async_b128(unsigned lds_off, const void* gaddr) {
    asm volatile("global_load_async_to_lds_b128 %0, %1, off"
                 :: "v"(lds_off), "v"((unsigned long long)(size_t)gaddr)
                 : "memory");
}
__device__ __forceinline__ void wait_async0() {
    asm volatile("s_wait_asynccnt 0x0" ::: "memory");
}
__device__ __forceinline__ void wait_async8() {   // tolerate 8 in-flight (next buffer)
    asm volatile("s_wait_asynccnt 0x8" ::: "memory");
}
__device__ __forceinline__ unsigned lds_addr(const void* p) {
    return (unsigned)(size_t)p;   // LDS aperture keeps offset in addr[31:0]
}

// =====================================================================
// Generic GEMM:  out[M,N] = act(A[M,K] @ W[K,N] + bias) (+ res)
//   aRowMod != 0  -> A row index = (global_row % aRowMod)  (broadcast A)
//   flags: 1 = ReLU, 2 = add residual (res indexed like out)
// Block: 128 threads (4 waves), tile 64x64, K-step 32, bf16 WMMA, f32 acc.
// Double-buffered f32 tiles staged via global_load_async_to_lds_b128;
// next tile's copies overlap current tile's WMMA work.
// =====================================================================
__global__ __launch_bounds__(128) void gemm_bf16(
    const float* __restrict__ A, const float* __restrict__ W,
    const float* __restrict__ bias, const float* __restrict__ res,
    float* __restrict__ out, int M, int K, int N, int aRowMod, int flags)
{
    __shared__ __attribute__((aligned(16))) float Asf[2][64][36]; // 144B pitch
    __shared__ __attribute__((aligned(16))) float Bsf[2][32][68]; // 272B pitch

    const int tid  = threadIdx.x;
    const int wave = tid >> 5, lane = tid & 31;
    const int half = lane >> 4, l15 = lane & 15;
    const int mo = (wave >> 1) * 32, no = (wave & 1) * 32;
    const int bm = blockIdx.y * 64, bn = blockIdx.x * 64;
    const bool fullT = (bm + 64 <= M);          // K,N always tile-aligned here

    const unsigned aBase = lds_addr(&Asf[0][0][0]);
    const unsigned bBase = lds_addr(&Bsf[0][0][0]);

    v8f c[2][2];
    for (int i = 0; i < 2; ++i) for (int j = 0; j < 2; ++j) c[i][j] = (v8f){};

    // issue async copies for one K-step into buffer b (8 chunks per thread)
    auto issueTiles = [&](int kk, int b) {
        const unsigned aB = aBase + (unsigned)b * 9216u;   // 64*36*4
        const unsigned bB = bBase + (unsigned)b * 8704u;   // 32*68*4
        for (int i = tid; i < 512; i += 128) {             // A tile 64x32
            int r = i >> 3, sg = i & 7;
            int ar = aRowMod ? ((bm + r) % aRowMod) : (bm + r);
            async_b128(aB + (unsigned)r * 144u + (unsigned)sg * 16u,
                       A + (size_t)ar * K + kk + sg * 4);
        }
        for (int i = tid; i < 512; i += 128) {             // W tile 32x64
            int r = i >> 4, sg = i & 15;
            async_b128(bB + (unsigned)r * 272u + (unsigned)sg * 16u,
                       W + (size_t)(kk + r) * N + bn + sg * 4);
        }
    };
    // fragment build + 2x2 WMMA on buffer b
    auto computeTile = [&](int b) {
        Frag af[2], bfr[2];
        for (int s = 0; s < 2; ++s) {
            int mr = mo + s * 16 + l15;
            for (int j = 0; j < 8; ++j) {
                int k = ((j >> 2) << 4) + half * 8 + ((j & 3) << 1);
                float2 t = *(const float2*)&Asf[b][mr][k];
                af[s].u[j] = packbf(t.x, t.y);
            }
            int nr = no + s * 16 + l15;
            for (int j = 0; j < 8; ++j) {
                int k = ((j >> 2) << 4) + half * 8 + ((j & 3) << 1);
                bfr[s].u[j] = packbf(Bsf[b][k][nr], Bsf[b][k + 1][nr]);
            }
        }
        for (int si = 0; si < 2; ++si)
            for (int sj = 0; sj < 2; ++sj)
                c[si][sj] = wmma_bf(af[si], bfr[sj], c[si][sj]);
    };

    if (fullT) {
        issueTiles(0, 0);
        int buf = 0;
        for (int kk = 0; kk < K; kk += 32, buf ^= 1) {
            const bool more = (kk + 32 < K);
            if (more) issueTiles(kk + 32, buf ^ 1);   // prefetch next K-step
            if (more) wait_async8(); else wait_async0();
            __syncthreads();
            computeTile(buf);
            __syncthreads();
        }
    } else {
        for (int kk = 0; kk < K; kk += 32) {
            for (int i = tid; i < 64 * 32; i += 128) {
                int r = i >> 5, cc = i & 31;
                int gr = bm + r, gc = kk + cc;
                float v = 0.0f;
                if (gr < M && gc < K) {
                    int ar = aRowMod ? (gr % aRowMod) : gr;
                    v = A[(size_t)ar * K + gc];
                }
                Asf[0][r][cc] = v;
            }
            for (int i = tid; i < 32 * 64; i += 128) {
                int r = i >> 6, cc = i & 63;
                int gk = kk + r, gn = bn + cc;
                Bsf[0][r][cc] = (gk < K && gn < N) ? W[(size_t)gk * N + gn] : 0.0f;
            }
            __syncthreads();
            computeTile(0);
            __syncthreads();
        }
    }

    for (int si = 0; si < 2; ++si)
        for (int sj = 0; sj < 2; ++sj) {
            int gcol = bn + no + sj * 16 + l15;
            for (int j = 0; j < 8; ++j) {
                int grow = bm + mo + si * 16 + half * 8 + j;
                if (grow < M && gcol < N) {
                    float v = c[si][sj][j];
                    if (bias) v += bias[gcol];
                    if (flags & 1) v = fmaxf(v, 0.0f);
                    if (flags & 2) v += res[(size_t)grow * N + gcol];
                    out[(size_t)grow * N + gcol] = v;
                }
            }
        }
}

// =====================================================================
// Attention: per (batch, head, 16-query tile):
//   S = softmax(Qh Kh^T * scale); O = Qh + S @ Vh
// Q,O: [B,nq,dv]; K,V: [B,m,dv]; head slice = cols [h*dh, (h+1)*dh)
// Block: 128 threads (4 waves). m <= 1024, dh <= 256.
// V tiles double-buffered in LDS via async b128 copies.
// =====================================================================
__global__ __launch_bounds__(128) void attn_kernel(
    const float* __restrict__ Q, const float* __restrict__ Kb,
    const float* __restrict__ Vm, float* __restrict__ O,
    int nq, int m, int dv, int dh, float scale)
{
    __shared__ float S[16][NN];                                      // 64KB
    __shared__ __attribute__((aligned(16))) float Vs[2][32 * 260];   // pitch dh+4
    __shared__ float red[16][8];
    __shared__ float rstat[16];

    const int tid  = threadIdx.x;
    const int wave = tid >> 5, lane = tid & 31;
    const int half = lane >> 4, l15 = lane & 15;
    const size_t bz = blockIdx.z;
    const int head = blockIdx.y;
    const int hd = head * dh;
    const int qt = blockIdx.x;

    // ---- phase 1: scores -> S ----
    int qrow_l = qt * 16 + l15; if (qrow_l > nq - 1) qrow_l = nq - 1;
    const float* qbase = Q + ((size_t)bz * nq + qrow_l) * dv + hd;

    Frag aq[8];
    const int nkf = dh >> 5;                    // 32-wide K-fragments over dh
    for (int f = 0; f < nkf; ++f)
        for (int j = 0; j < 8; ++j) {
            int k = f * 32 + ((j >> 2) << 4) + half * 8 + ((j & 3) << 1);
            float2 t = *(const float2*)(qbase + k);
            aq[f].u[j] = packbf(t.x, t.y);
        }

    const int mtiles = m >> 4;
    for (int mt = wave; mt < mtiles; mt += 4) {
        v8f c = (v8f){};
        const float* kbase = Kb + ((size_t)bz * m + mt * 16 + l15) * dv + hd;
        for (int f = 0; f < nkf; ++f) {
            Frag bk;
            for (int j = 0; j < 8; ++j) {
                int k = f * 32 + ((j >> 2) << 4) + half * 8 + ((j & 3) << 1);
                float2 t = *(const float2*)(kbase + k);
                bk.u[j] = packbf(t.x, t.y);
            }
            c = wmma_bf(aq[f], bk, c);
        }
        for (int j = 0; j < 8; ++j)
            S[half * 8 + j][mt * 16 + l15] = c[j] * scale;
    }
    __syncthreads();

    // ---- phase 2: row softmax over m ----
    {
        const int row = tid >> 3, part = tid & 7;
        float mx = -3.4e38f;
        for (int cidx = part; cidx < m; cidx += 8) mx = fmaxf(mx, S[row][cidx]);
        red[row][part] = mx; __syncthreads();
        if (part == 0) {
            float v = red[row][0];
            for (int p = 1; p < 8; ++p) v = fmaxf(v, red[row][p]);
            rstat[row] = v;
        }
        __syncthreads();
        float rm = rstat[row], sum = 0.0f;
        for (int cidx = part; cidx < m; cidx += 8) {
            float e = __expf(S[row][cidx] - rm);
            S[row][cidx] = e; sum += e;
        }
        red[row][part] = sum; __syncthreads();
        if (part == 0) {
            float v = 0.0f;
            for (int p = 0; p < 8; ++p) v += red[row][p];
            rstat[row] = v;
        }
        __syncthreads();
        float inv = 1.0f / rstat[row];
        for (int cidx = part; cidx < m; cidx += 8) S[row][cidx] *= inv;
    }
    __syncthreads();

    // ---- phase 3: O = Qh + S @ Vh (double-buffered V tiles) ----
    const int ntiles = dh >> 4;         // 16 (dh=256) or 8 (dh=128)
    const int myNt   = ntiles >> 2;     // per-wave n-tiles
    const int vpitch = dh + 4;          // floats; pitch*4 is 16B multiple
    const unsigned vBase = lds_addr(&Vs[0][0]);
    v8f co[4];
    for (int t = 0; t < myNt; ++t) co[t] = (v8f){};

    auto issueV = [&](int mm, int b) {      // dh/16 chunks per thread (8 or 16)
        const unsigned base = vBase + (unsigned)b * (32u * 260u * 4u);
        const int segs = dh >> 2;
        for (int i = tid; i < 32 * segs; i += 128) {
            int r = i / segs, sg = i - r * segs;
            async_b128(base + (unsigned)(r * vpitch + sg * 4) * 4u,
                       Vm + ((size_t)bz * m + mm + r) * dv + hd + sg * 4);
        }
    };

    issueV(0, 0);
    int buf = 0;
    for (int mm = 0; mm < m; mm += 32, buf ^= 1) {
        const bool more = (mm + 32 < m);
        if (more) issueV(mm + 32, buf ^ 1);
        if (more) wait_async8(); else wait_async0();  // in-order: current resident
        __syncthreads();

        const float* Vb = &Vs[buf][0];
        Frag af;
        for (int j = 0; j < 8; ++j) {
            int k = mm + ((j >> 2) << 4) + half * 8 + ((j & 3) << 1);
            af.u[j] = packbf(S[l15][k], S[l15][k + 1]);
        }
        for (int t = 0; t < myNt; ++t) {
            int nb = (wave * myNt + t) * 16 + l15;
            Frag bv;
            for (int j = 0; j < 8; ++j) {
                int kr = ((j >> 2) << 4) + half * 8 + ((j & 3) << 1);
                bv.u[j] = packbf(Vb[kr * vpitch + nb], Vb[(kr + 1) * vpitch + nb]);
            }
            co[t] = wmma_bf(af, bv, co[t]);
        }
        __syncthreads();
    }

    for (int t = 0; t < myNt; ++t) {
        int col = (wave * myNt + t) * 16 + l15;
        for (int j = 0; j < 8; ++j) {
            int qr = qt * 16 + half * 8 + j;
            if (qr < nq) {
                size_t idx = ((size_t)bz * nq + qr) * dv + hd + col;
                O[idx] = Q[idx] + co[t][j];
            }
        }
    }
}

// =====================================================================
// LayerNorm per row: out = (x-mean)*rsqrt(var+eps)*g + be (+ res)
// =====================================================================
__global__ __launch_bounds__(256) void ln_kernel(
    const float* __restrict__ in, const float* __restrict__ g,
    const float* __restrict__ be, const float* __restrict__ res,
    float* __restrict__ out, int D)
{
    const size_t row = blockIdx.x;
    const float* x = in + row * D;
    __shared__ float s1[256], s2[256];
    float a = 0.0f, b = 0.0f;
    for (int i = threadIdx.x; i < D; i += 256) {
        float v = x[i]; a += v; b += v * v;
    }
    s1[threadIdx.x] = a; s2[threadIdx.x] = b; __syncthreads();
    for (int o = 128; o > 0; o >>= 1) {
        if ((int)threadIdx.x < o) {
            s1[threadIdx.x] += s1[threadIdx.x + o];
            s2[threadIdx.x] += s2[threadIdx.x + o];
        }
        __syncthreads();
    }
    float mean = s1[0] / (float)D;
    float var  = s2[0] / (float)D - mean * mean;
    float inv  = rsqrtf(var + 1e-5f);
    for (int i = threadIdx.x; i < D; i += 256) {
        float v = (x[i] - mean) * inv * g[i] + be[i];
        if (res) v += res[row * D + i];
        out[row * D + i] = v;
    }
}

// =====================================================================
// Classifier: logits = x[8,DL] @ W[DL,NC] + b ; probs = softmax(logits)
// Writes logits to out[0:144], probs to out[144:288].
// =====================================================================
__global__ __launch_bounds__(256) void cls_kernel(
    const float* __restrict__ x, const float* __restrict__ W,
    const float* __restrict__ bias, float* __restrict__ out)
{
    __shared__ float lg[BB][NC_];
    int t = threadIdx.x;
    if (t < BB * NC_) {
        int bb = t / NC_, c = t % NC_;
        float acc = bias[c];
        for (int k = 0; k < DL_; ++k) acc += x[bb * DL_ + k] * W[k * NC_ + c];
        lg[bb][c] = acc;
        out[bb * NC_ + c] = acc;
    }
    __syncthreads();
    if (t < BB) {
        float mx = -3.4e38f;
        for (int c = 0; c < NC_; ++c) mx = fmaxf(mx, lg[t][c]);
        float s = 0.0f, e[NC_];
        for (int c = 0; c < NC_; ++c) { e[c] = __expf(lg[t][c] - mx); s += e[c]; }
        for (int c = 0; c < NC_; ++c) out[BB * NC_ + t * NC_ + c] = e[c] / s;
    }
}

// ============================ host side ==============================
struct Mab {
    const float *Wq,*bq,*Wk,*bk,*Wv,*bv,*Wo,*bo,*g0,*be0,*g1,*be1;
};
static Mab unpackMab(void* const* d_in, int base) {
    Mab m;
    m.Wq = (const float*)d_in[base + 0];  m.bq  = (const float*)d_in[base + 1];
    m.Wk = (const float*)d_in[base + 2];  m.bk  = (const float*)d_in[base + 3];
    m.Wv = (const float*)d_in[base + 4];  m.bv  = (const float*)d_in[base + 5];
    m.Wo = (const float*)d_in[base + 6];  m.bo  = (const float*)d_in[base + 7];
    m.g0 = (const float*)d_in[base + 8];  m.be0 = (const float*)d_in[base + 9];
    m.g1 = (const float*)d_in[base + 10]; m.be1 = (const float*)d_in[base + 11];
    return m;
}

static void gemm(hipStream_t s, const float* A, const float* W, const float* bias,
                 const float* res, float* out, int M, int K, int N,
                 int aRowMod, int flags) {
    dim3 g((N + 63) / 64, (M + 63) / 64);
    gemm_bf16<<<g, dim3(128), 0, s>>>(A, W, bias, res, out, M, K, N, aRowMod, flags);
}
static void attn(hipStream_t s, const float* Q, const float* K, const float* V,
                 float* O, int nq, int m, int dv, float scale) {
    dim3 g((nq + 15) / 16, HH, BB);
    attn_kernel<<<g, dim3(128), 0, s>>>(Q, K, V, O, nq, m, dv, dv / HH, scale);
}
static void lnorm(hipStream_t s, const float* in, const float* g, const float* be,
                  const float* res, float* out, int M, int D) {
    ln_kernel<<<dim3(M), dim3(256), 0, s>>>(in, g, be, res, out, D);
}

extern "C" void kernel_launch(void* const* d_in, const int* in_sizes, int n_in,
                              void* d_out, int out_size, void* d_ws, size_t ws_size,
                              hipStream_t stream)
{
    (void)in_sizes; (void)n_in; (void)out_size; (void)ws_size;
    // ---- unpack inputs (setup_inputs insertion order, recursive) ----
    const float* X   = (const float*)d_in[0];
    const float* e1W = (const float*)d_in[1];
    const float* e1b = (const float*)d_in[2];
    const float* I2  = (const float*)d_in[3];
    Mab e2m0 = unpackMab(d_in, 4);
    Mab e2m1 = unpackMab(d_in, 16);
    const float* I3  = (const float*)d_in[28];
    Mab e3m0 = unpackMab(d_in, 29);
    Mab e3m1 = unpackMab(d_in, 41);
    const float* pS  = (const float*)d_in[53];
    Mab pm   = unpackMab(d_in, 54);
    const float* cW  = (const float*)d_in[66];
    const float* cb  = (const float*)d_in[67];

    // ---- workspace arena (floats): 4 x 64MB + 16MB ----
    float* wsf = (float*)d_ws;
    const size_t XSZ = (size_t)BB * NN * DH_;   // 16,777,216
    const size_t HSZ = (size_t)BB * NI_ * DH_;  //  4,194,304
    float* X0 = wsf;
    float* G1 = wsf + XSZ;
    float* G2 = wsf + 2 * XSZ;
    float* G3 = wsf + 3 * XSZ;
    float* S4 = wsf + 4 * XSZ;                  // HSZ floats

    const float scE = 1.0f / sqrtf((float)DH_);
    const float scP = 1.0f / sqrtf((float)DL_);

    // enc1: x = X @ W + b
    gemm(stream, X, e1W, e1b, nullptr, X0, BB * NN, DIN_, DH_, 0, 0);

    auto isab = [&](const float* I, const Mab& m0, const Mab& m1) {
        // ---- mab0(I, x): H ----
        gemm(stream, I,  m0.Wq, m0.bq, nullptr, G1, BB * NI_, DH_, DH_, NI_, 0); // Q0
        gemm(stream, X0, m0.Wk, m0.bk, nullptr, G2, BB * NN,  DH_, DH_, 0, 0);   // K0
        gemm(stream, X0, m0.Wv, m0.bv, nullptr, G3, BB * NN,  DH_, DH_, 0, 0);   // V0
        attn(stream, G1, G2, G3, S4, NI_, NN, DH_, scE);                          // O0
        lnorm(stream, S4, m0.g0, m0.be0, nullptr, S4, BB * NI_, DH_);
        gemm(stream, S4, m0.Wo, m0.bo, S4, G1, BB * NI_, DH_, DH_, 0, 3);        // +relu+res
        lnorm(stream, G1, m0.g1, m0.be1, nullptr, S4, BB * NI_, DH_);            // H -> S4
        // ---- mab1(x, H) + x ----
        gemm(stream, X0, m1.Wq, m1.bq, nullptr, G1, BB * NN,  DH_, DH_, 0, 0);   // Q1
        gemm(stream, S4, m1.Wk, m1.bk, nullptr, G2, BB * NI_, DH_, DH_, 0, 0);   // K1
        gemm(stream, S4, m1.Wv, m1.bv, nullptr, G2 + HSZ, BB * NI_, DH_, DH_, 0, 0); // V1
        attn(stream, G1, G2, G2 + HSZ, G3, NN, NI_, DH_, scE);                   // O1
        lnorm(stream, G3, m1.g0, m1.be0, nullptr, G3, BB * NN, DH_);
        gemm(stream, G3, m1.Wo, m1.bo, G3, G1, BB * NN, DH_, DH_, 0, 3);
        lnorm(stream, G1, m1.g1, m1.be1, X0, X0, BB * NN, DH_);                  // x = LN + x
    };
    isab(I2, e2m0, e2m1);
    isab(I3, e3m0, e3m1);

    // ---- PMA: mab(S, x), NS=1 ----
    const size_t KVSZ = (size_t)BB * NN * DL_;  // 8,388,608
    gemm(stream, pS, pm.Wq, pm.bq, nullptr, G1, BB * 1, DL_, DL_, 1, 0);         // Q
    gemm(stream, X0, pm.Wk, pm.bk, nullptr, G2, BB * NN, DH_, DL_, 0, 0);        // K
    gemm(stream, X0, pm.Wv, pm.bv, nullptr, G2 + KVSZ, BB * NN, DH_, DL_, 0, 0); // V
    attn(stream, G1, G2, G2 + KVSZ, G3, 1, NN, DL_, scP);
    lnorm(stream, G3, pm.g0, pm.be0, nullptr, G3, BB, DL_);
    gemm(stream, G3, pm.Wo, pm.bo, G3, G1, BB, DL_, DL_, 0, 3);
    lnorm(stream, G1, pm.g1, pm.be1, nullptr, G3, BB, DL_);

    // ---- classifier + softmax -> d_out ----
    cls_kernel<<<dim3(1), dim3(256), 0, stream>>>(G3, cW, cb, (float*)d_out);
}
